// RSNN_49082886259324
// MI455X (gfx1250) — matrix operand order repaired and checked by reading, not compiled, not run on hardware
//
#include <hip/hip_runtime.h>
#include <math.h>

// ---------------------------------------------------------------------------
// ALIF spiking RNN forward for MI455X (gfx1250, wave32).
//  * GEMMs: V_WMMA_F32_16X16X32_BF16 (spikes exact in bf16, f32 accumulate).
//  * A operands (spike matrices) DMA'd into LDS by the Tensor Data Mover
//    (TENSOR_LOAD_TO_LDS, TENSORcnt) and shared by all waves of a block;
//    fragments then come from LDS (ds_load_b128) while bf16 weights stream
//    from the 192MB L2, double-buffered so WMMAs overlap in-flight loads.
//  * LDS tile is dynamic (extern __shared__): the TDM writes LDS behind the
//    compiler's back, and a static non-escaping __shared__ object's loads
//    get folded to undef. An external AS(3) global cannot be proven
//    unwritten, so the ds_loads must be emitted.
//  * Roofline: ~94 GFLOP of GEMM, 524 MB of output stores (~22.5us floor
//    at 23.3 TB/s); weights (6.3 MB bf16) stay L2-resident for all 100 steps.
// ---------------------------------------------------------------------------

typedef __attribute__((ext_vector_type(16))) __bf16       v16bf;
typedef __attribute__((ext_vector_type(8)))  __bf16       v8bf;
typedef __attribute__((ext_vector_type(8)))  float        v8f;
typedef __attribute__((ext_vector_type(4)))  unsigned int v4u;
typedef __attribute__((ext_vector_type(8)))  int          v8i;
typedef __attribute__((ext_vector_type(4)))  int          v4i;

#define T_STEPS 100
#define BATCH   256
#define NN      512
#define BN      (BATCH * NN)          // 131072
#define OUT_SZ  (T_STEPS * BN)

#define B0f   0.01f
#define BETAf 1.8f

__device__ __forceinline__ void mem_clobber() { asm volatile("" ::: "memory"); }

// ---- Tensor Data Mover: 2D bf16 tile (rows x 512) global -> LDS -----------
// D# group0: count=1 | lds_addr | global_addr[56:0] | type=2
// D# group1: data_size=2B | tensor_dim0/1 | tile_dim0/1 | dim0_stride
__device__ __forceinline__ void tdm_load_tile_bf16(const __bf16* gsrc,
                                                   unsigned lds_byte_off,
                                                   unsigned rows) {
    unsigned long long ga = (unsigned long long)gsrc;
    v4u g0;
    g0[0] = 1u;                                          // count=1
    g0[1] = lds_byte_off;                                // lds_addr (bytes)
    g0[2] = (unsigned)(ga & 0xffffffffu);                // global_addr[31:0]
    g0[3] = (unsigned)((ga >> 32) & 0x01ffffffu)         // global_addr[56:32]
          | (2u << 30);                                  // type=2 (image)
    unsigned td0 = NN, td1 = rows, tile0 = NN, tile1 = rows;
    unsigned long long stride0 = NN;                     // data_size units
    v8i g1;
    g1[0] = (int)(1u << 16);                             // data_size=1 -> 2B
    g1[1] = (int)((td0 & 0xffffu) << 16);                // tensor_dim0[15:0]
    g1[2] = (int)(((td0 >> 16) & 0xffffu) | ((td1 & 0xffffu) << 16));
    g1[3] = (int)(((td1 >> 16) & 0xffffu) | ((tile0 & 0xffffu) << 16));
    g1[4] = (int)(tile1 & 0xffffu);                      // tile_dim1 (tile_dim2=0)
    g1[5] = (int)(unsigned)(stride0 & 0xffffffffu);      // dim0_stride[31:0]
    g1[6] = (int)(unsigned)((stride0 >> 32) & 0xffffu);  // dim0_stride[47:32]
    g1[7] = 0;
    v4i z = {0, 0, 0, 0};
#if __clang_major__ >= 23
    v8i z8 = {0, 0, 0, 0, 0, 0, 0, 0};
    __builtin_amdgcn_tensor_load_to_lds(g0, g1, z, z, z8, 0);
#else
    __builtin_amdgcn_tensor_load_to_lds(g0, g1, z, z, 0);
#endif
    mem_clobber();
}

// Never-executed at runtime (blockDim.y == 1), but the compiler cannot prove
// it: a may-store with an opaque index that aliases every LDS load, so loads
// from the tile cannot be folded away.
__device__ __forceinline__ void lds_fake_store(__bf16* sm) {
    if (__builtin_expect(blockDim.y > 1, 0))
        sm[threadIdx.x * blockDim.y] = (__bf16)0.0f;
}

// ---- WMMA fragment loaders -------------------------------------------------
// A matrix 16x32 tile from LDS (row-major 16 x NN tile). ISA A layout:
// lanes 0-15: M=lane, K={0..7,16..23}; lanes 16-31: M=lane-16, K={8..15,24..31}
__device__ __forceinline__ v16bf load_frag_a_lds(const __bf16* sm, int kb) {
    int lane = threadIdx.x & 31;
    int r = lane & 15, h = (lane >> 4) & 1;
    const __bf16* p = sm + r * NN + kb + h * 8;
    v8bf lo = *(const v8bf*)(p);
    v8bf hi = *(const v8bf*)(p + 16);
    v16bf o;
#pragma unroll
    for (int i = 0; i < 8; ++i) { o[i] = lo[i]; o[i + 8] = hi[i]; }
    return o;
}

// B matrix 32x16 tile; column n of B = row n of W (C = X @ W^T). ISA B layout:
// lanes 0-15 hold K={0..7,16..23} for N=lane, lanes 16-31 hold K={8..15,24..31}.
__device__ __forceinline__ v16bf load_frag_b(const __bf16* __restrict__ wrow,
                                             int col0, int kb) {
    int lane = threadIdx.x & 31;
    int r = lane & 15, h = (lane >> 4) & 1;
    const __bf16* p = wrow + (size_t)(col0 + r) * NN + kb + h * 8;
    v8bf lo = *(const v8bf*)(p);
    v8bf hi = *(const v8bf*)(p + 16);
    v16bf o;
#pragma unroll
    for (int i = 0; i < 8; ++i) { o[i] = lo[i]; o[i + 8] = hi[i]; }
    return o;
}

// One wave accumulates C[16, n0:n0+64] += smA(16xNN) @ W^T, K double-buffered
// so next-iteration LDS/global loads are in flight during current WMMAs.
__device__ __forceinline__ void gemm_lds_a(const __bf16* sm,
                                           const __bf16* __restrict__ W,
                                           int n0, v8f acc[4]) {
    v16bf a = load_frag_a_lds(sm, 0);
    v16bf b[4];
#pragma unroll
    for (int j = 0; j < 4; ++j) b[j] = load_frag_b(W, n0 + 16 * j, 0);
    for (int kb = 0; kb < NN - 32; kb += 32) {
        v16bf an = load_frag_a_lds(sm, kb + 32);
        v16bf bn[4];
#pragma unroll
        for (int j = 0; j < 4; ++j) bn[j] = load_frag_b(W, n0 + 16 * j, kb + 32);
#pragma unroll
        for (int j = 0; j < 4; ++j)
            acc[j] = __builtin_amdgcn_wmma_f32_16x16x32_bf16(
                false, a, false, b[j], (short)0, acc[j], false, false);
        a = an;
#pragma unroll
        for (int j = 0; j < 4; ++j) b[j] = bn[j];
    }
#pragma unroll
    for (int j = 0; j < 4; ++j)
        acc[j] = __builtin_amdgcn_wmma_f32_16x16x32_bf16(
            false, a, false, b[j], (short)0, acc[j], false, false);
}

// ---- prep: bf16 weight conversion, decay precompute, state init -----------
__global__ void k_prep(const float* __restrict__ enc_W, const float* __restrict__ Wd,
                       const float* __restrict__ Wr, const float* __restrict__ W_out,
                       const float* __restrict__ p0,
                       const float* __restrict__ tau_m_in, const float* __restrict__ tau_adp_in,
                       const float* __restrict__ tau_m_rnn, const float* __restrict__ tau_adp_rnn,
                       const float* __restrict__ tau_m_out,
                       __bf16* encW_bf, __bf16* Wd_bf, __bf16* Wr_bf, __bf16* Wout_bf,
                       __bf16* p0s_bf, __bf16* spr_bf /*[3][2][BN]*/,
                       float* sp_in, float* ad_in, float* ad_r /*[3][BN]*/,
                       float* dec /*[9][NN]*/) {
    int i = blockIdx.x * blockDim.x + threadIdx.x;
    if (i < 5 * NN * NN) encW_bf[i] = (__bf16)enc_W[i];
    if (i < 3 * NN * NN) { Wd_bf[i] = (__bf16)Wd[i]; Wr_bf[i] = (__bf16)Wr[i]; }
    if (i < NN * NN) Wout_bf[i] = (__bf16)W_out[i];
    if (i < BN) {
        p0s_bf[i] = (__bf16)(p0[i] * 1000.0f);
        sp_in[i] = 0.0f;
        ad_in[i] = B0f;
#pragma unroll
        for (int l = 0; l < 3; ++l) {
            spr_bf[(size_t)(l * 2) * BN + i] = (__bf16)0.0f;  // parity-0 = "old" at t=0
            ad_r[(size_t)l * BN + i] = B0f;
        }
    }
    if (i < NN) {
        dec[0 * NN + i] = __expf(-1.0f / tau_m_in[i]);
        dec[1 * NN + i] = __expf(-1.0f / tau_adp_in[i]);
#pragma unroll
        for (int l = 0; l < 3; ++l) {
            dec[(2 + l) * NN + i] = __expf(-1.0f / tau_m_rnn[l * NN + i]);
            dec[(5 + l) * NN + i] = __expf(-1.0f / tau_adp_rnn[l * NN + i]);
        }
        dec[8 * NN + i] = __expf(-1.0f / tau_m_out[i]);
    }
}

// ---- init membranes: 5x [256x512]@[512x512]^T, sigmoid epilogue -----------
// grid: 5 layers x 16 m-tiles x 2 n-groups = 160 blocks of 128 (4 waves)
// dynamic LDS: 16*NN*2 bytes
__global__ void k_init_mem(const __bf16* __restrict__ p0s_bf,
                           const __bf16* __restrict__ encW_bf,
                           const float* __restrict__ enc_b,
                           float* mem_in, float* mem_r /*[3][BN]*/, float* mem_out) {
    extern __shared__ __bf16 smdyn[];
    __bf16* smA = smdyn;
    int l = blockIdx.x >> 5;
    int rblk = blockIdx.x & 31;
    int m0 = (rblk >> 1) * 16;
    int wave = threadIdx.x >> 5;
    int n0 = (rblk & 1) * 256 + wave * 64;

    lds_fake_store(smA);
    if (wave == 0) {
        tdm_load_tile_bf16(p0s_bf + (size_t)m0 * NN, 0u, 16u);
        __builtin_amdgcn_s_wait_tensorcnt(0);
    }
    __syncthreads();
    mem_clobber();

    v8f acc[4] = {};
    gemm_lds_a(smA, encW_bf + (size_t)l * NN * NN, n0, acc);

    float* dst = (l == 0) ? mem_in : ((l == 4) ? mem_out : (mem_r + (size_t)(l - 1) * BN));
    int lane = threadIdx.x & 31;
    int h = lane >> 4, nc = lane & 15;
#pragma unroll
    for (int j = 0; j < 4; ++j) {
        int n = n0 + j * 16 + nc;
        float eb = enc_b[l * NN + n];
#pragma unroll
        for (int e = 0; e < 8; ++e) {
            int b = m0 + h * 8 + e;
            float pre = acc[j][e] + eb;
            float sig = 1.0f / (1.0f + __expf(-pre));
            dst[(size_t)b * NN + n] = 1.5f * B0f * sig;   // ratio * B0 * sigmoid
        }
    }
}

// ---- input layer: K=2 matmul + ALIF (elementwise) -------------------------
__global__ void k_in(const float* __restrict__ v, const float* __restrict__ W_in,
                     const float* __restrict__ b_in,
                     const float* __restrict__ a_in, const float* __restrict__ r_in,
                     float* mem_st, float* sp_st, float* ad_st,
                     __bf16* x0_bf, float* out_mem, float* out_sp, int t) {
    int i = blockIdx.x * blockDim.x + threadIdx.x;   // BN threads
    int b = i >> 9, n = i & (NN - 1);
    const float* vt = v + (size_t)t * BATCH * 2 + b * 2;
    float inp = vt[0] * W_in[n * 2] + vt[1] * W_in[n * 2 + 1] + b_in[n];
    float al = a_in[n], ro = r_in[n];
    float spk = sp_st[i];
    float ad = ro * ad_st[i] + (1.0f - ro) * spk;
    float Bth = B0f + BETAf * ad;
    float mem = mem_st[i] * al + (1.0f - al) * inp - Bth * spk;
    float ns = (mem - Bth) > 0.0f ? 1.0f : 0.0f;
    mem_st[i] = mem; sp_st[i] = ns; ad_st[i] = ad;
    x0_bf[i] = (__bf16)ns;
    out_mem[(size_t)t * BN + i] = mem;
    out_sp[(size_t)t * BN + i] = ns;
}

// ---- recurrent layer: X@Wd^T + S@Wr^T, fused ALIF epilogue ----------------
// grid: 16 m-tiles x 2 n-groups = 32 blocks of 128 (4 waves).
// dynamic LDS: 2*16*NN*2 bytes. TDM stages both A tiles into LDS.
__global__ void k_rec(const __bf16* __restrict__ xin, const __bf16* __restrict__ spr_old,
                      __bf16* spr_new,
                      const __bf16* __restrict__ Wd_bf, const __bf16* __restrict__ Wr_bf,
                      const float* __restrict__ bd, const float* __restrict__ br,
                      const float* __restrict__ a_vec, const float* __restrict__ r_vec,
                      float* mem_st, float* ad_st,
                      float* out_m, float* out_s, int t) {
    extern __shared__ __bf16 smdyn[];
    __bf16* smX = smdyn;               // x tile
    __bf16* smS = smdyn + 16 * NN;     // old-spike tile

    int m0 = (int)(blockIdx.x >> 1) * 16;
    int wave = threadIdx.x >> 5;
    int n0 = (int)(blockIdx.x & 1) * 256 + wave * 64;

    lds_fake_store(smX);
    if (wave == 0) {
        tdm_load_tile_bf16(xin + (size_t)m0 * NN, 0u, 16u);
        tdm_load_tile_bf16(spr_old + (size_t)m0 * NN, (unsigned)(16 * NN * 2), 16u);
        __builtin_amdgcn_s_wait_tensorcnt(0);
    }
    __syncthreads();
    mem_clobber();

    v8f acc[4] = {};
    gemm_lds_a(smX, Wd_bf, n0, acc);       // feedforward term
    gemm_lds_a(smS, Wr_bf, n0, acc);       // recurrent term

    int lane = threadIdx.x & 31;
    int h = lane >> 4, nc = lane & 15;
#pragma unroll
    for (int j = 0; j < 4; ++j) {
        int n = n0 + j * 16 + nc;
        float bias = bd[n] + br[n];
        float al = a_vec[n], ro = r_vec[n];
#pragma unroll
        for (int e = 0; e < 8; ++e) {
            int row = h * 8 + e;
            int b = m0 + row;
            size_t idx = (size_t)b * NN + n;
            float pre = acc[j][e] + bias;
            float spk = (float)smS[row * NN + n];      // old spike (exact in bf16)
            float ad = ro * ad_st[idx] + (1.0f - ro) * spk;
            float Bth = B0f + BETAf * ad;
            float mem = mem_st[idx] * al + (1.0f - al) * pre - Bth * spk;
            float ns = (mem - Bth) > 0.0f ? 1.0f : 0.0f;
            mem_st[idx] = mem; ad_st[idx] = ad;
            spr_new[idx] = (__bf16)ns;
            size_t o = (size_t)t * BN + idx;
            out_m[o] = mem;
            out_s[o] = ns;
        }
    }
}

// ---- readout: X@W_out^T, leaky integrator + relu --------------------------
// dynamic LDS: 16*NN*2 bytes
__global__ void k_out(const __bf16* __restrict__ xin, const __bf16* __restrict__ Wout_bf,
                      const float* __restrict__ b_out, const float* __restrict__ a_out,
                      float* mem_out_st, float* out0, float* out9, int t) {
    extern __shared__ __bf16 smdyn[];
    __bf16* smA = smdyn;
    int m0 = (int)(blockIdx.x >> 1) * 16;
    int wave = threadIdx.x >> 5;
    int n0 = (int)(blockIdx.x & 1) * 256 + wave * 64;

    lds_fake_store(smA);
    if (wave == 0) {
        tdm_load_tile_bf16(xin + (size_t)m0 * NN, 0u, 16u);
        __builtin_amdgcn_s_wait_tensorcnt(0);
    }
    __syncthreads();
    mem_clobber();

    v8f acc[4] = {};
    gemm_lds_a(smA, Wout_bf, n0, acc);

    int lane = threadIdx.x & 31;
    int h = lane >> 4, nc = lane & 15;
#pragma unroll
    for (int j = 0; j < 4; ++j) {
        int n = n0 + j * 16 + nc;
        float bias = b_out[n];
        float ao = a_out[n];
#pragma unroll
        for (int e = 0; e < 8; ++e) {
            int b = m0 + h * 8 + e;
            size_t idx = (size_t)b * NN + n;
            float pre = acc[j][e] + bias;
            float mem = mem_out_st[idx] * ao + (1.0f - ao) * pre;
            mem_out_st[idx] = mem;
            float g = mem > 0.0f ? mem : 0.0f;
            size_t o = (size_t)t * BN + idx;
            out0[o] = g;
            out9[o] = g;
        }
    }
}

// ---------------------------------------------------------------------------
extern "C" void kernel_launch(void* const* d_in, const int* in_sizes, int n_in,
                              void* d_out, int out_size, void* d_ws, size_t ws_size,
                              hipStream_t stream) {
    const float* v          = (const float*)d_in[0];   // [T,B,2]
    const float* p0         = (const float*)d_in[1];   // [B,512]
    const float* enc_W      = (const float*)d_in[2];   // [5,N,N]
    const float* enc_b      = (const float*)d_in[3];   // [5,N]
    const float* W_in       = (const float*)d_in[4];   // [N,2]
    const float* b_in       = (const float*)d_in[5];   // [N]
    const float* tau_m_in   = (const float*)d_in[6];
    const float* tau_adp_in = (const float*)d_in[7];
    const float* Wd         = (const float*)d_in[8];   // [3,N,N]
    const float* bd         = (const float*)d_in[9];   // [3,N]
    const float* Wr         = (const float*)d_in[10];  // [3,N,N]
    const float* br         = (const float*)d_in[11];  // [3,N]
    const float* tau_m_rnn  = (const float*)d_in[12];  // [3,N]
    const float* tau_adp_rnn= (const float*)d_in[13];  // [3,N]
    const float* W_out      = (const float*)d_in[14];  // [N,N]
    const float* b_out      = (const float*)d_in[15];  // [N]
    const float* tau_m_out  = (const float*)d_in[16];  // [N]

    // ---- workspace carve-up (256B aligned) --------------------------------
    char* w = (char*)d_ws;
    auto carve = [&](size_t bytes) -> char* {
        char* p = w;
        w += (bytes + 255) & ~(size_t)255;
        return p;
    };
    __bf16* encW_bf = (__bf16*)carve((size_t)5 * NN * NN * 2);
    __bf16* Wd_bf   = (__bf16*)carve((size_t)3 * NN * NN * 2);
    __bf16* Wr_bf   = (__bf16*)carve((size_t)3 * NN * NN * 2);
    __bf16* Wout_bf = (__bf16*)carve((size_t)NN * NN * 2);
    __bf16* p0s_bf  = (__bf16*)carve((size_t)BN * 2);
    __bf16* x0_bf   = (__bf16*)carve((size_t)BN * 2);
    __bf16* spr_bf  = (__bf16*)carve((size_t)3 * 2 * BN * 2);  // [layer][parity][BN]
    float*  mem_in  = (float*)carve((size_t)BN * 4);
    float*  sp_in   = (float*)carve((size_t)BN * 4);
    float*  ad_in   = (float*)carve((size_t)BN * 4);
    float*  mem_r   = (float*)carve((size_t)3 * BN * 4);
    float*  ad_r    = (float*)carve((size_t)3 * BN * 4);
    float*  mem_out = (float*)carve((size_t)BN * 4);
    float*  dec     = (float*)carve((size_t)9 * NN * 4);   // a_in,r_in,a_rnn[3],r_rnn[3],a_out

    float* out = (float*)d_out;   // 10 arrays of [T,B,N]
    float* o_gout  = out + (size_t)0 * OUT_SZ;
    float* o_memin = out + (size_t)1 * OUT_SZ;
    float* o_spin  = out + (size_t)2 * OUT_SZ;
    float* o_gout2 = out + (size_t)9 * OUT_SZ;

    const size_t lds1 = (size_t)16 * NN * 2;       // 16 KB: one A tile
    const size_t lds2 = (size_t)2 * 16 * NN * 2;   // 32 KB: two A tiles

    // ---- prep + init membranes -------------------------------------------
    {
        int total = 5 * NN * NN;
        k_prep<<<(total + 255) / 256, 256, 0, stream>>>(
            enc_W, Wd, Wr, W_out, p0, tau_m_in, tau_adp_in,
            tau_m_rnn, tau_adp_rnn, tau_m_out,
            encW_bf, Wd_bf, Wr_bf, Wout_bf, p0s_bf, spr_bf,
            sp_in, ad_in, ad_r, dec);
    }
    k_init_mem<<<160, 128, lds1, stream>>>(p0s_bf, encW_bf, enc_b, mem_in, mem_r, mem_out);

    // ---- time loop --------------------------------------------------------
    for (int t = 0; t < T_STEPS; ++t) {
        k_in<<<BN / 256, 256, 0, stream>>>(v, W_in, b_in, dec + 0 * NN, dec + 1 * NN,
                                           mem_in, sp_in, ad_in, x0_bf,
                                           o_memin, o_spin, t);
        int po = t & 1, pn = (t + 1) & 1;   // ping-pong spike buffers
        for (int l = 0; l < 3; ++l) {
            const __bf16* xin = (l == 0) ? (const __bf16*)x0_bf
                                         : spr_bf + (size_t)((l - 1) * 2 + pn) * BN;
            const __bf16* sold = spr_bf + (size_t)(l * 2 + po) * BN;
            __bf16*       snew = spr_bf + (size_t)(l * 2 + pn) * BN;
            float* om = out + (size_t)(3 + 2 * l) * OUT_SZ;   // m1,m2,m3
            float* os = out + (size_t)(4 + 2 * l) * OUT_SZ;   // s1,s2,s3
            k_rec<<<32, 128, lds2, stream>>>(xin, sold, snew,
                                             Wd_bf + (size_t)l * NN * NN,
                                             Wr_bf + (size_t)l * NN * NN,
                                             bd + l * NN, br + l * NN,
                                             dec + (2 + l) * NN, dec + (5 + l) * NN,
                                             mem_r + (size_t)l * BN, ad_r + (size_t)l * BN,
                                             om, os, t);
        }
        const __bf16* x3 = spr_bf + (size_t)(2 * 2 + pn) * BN;
        k_out<<<32, 128, lds1, stream>>>(x3, Wout_bf, b_out, dec + 8 * NN,
                                         mem_out, o_gout, o_gout2, t);
    }
}
